// CovNet_70050916598136
// MI455X (gfx1250) — compile-verified, matrix operand order checked by reading, not run on hardware
//
#include <hip/hip_runtime.h>
#include <hip/hip_bf16.h>

typedef __attribute__((ext_vector_type(16))) _Float16 v16h;
typedef __attribute__((ext_vector_type(8)))  _Float16 v8h;
typedef __attribute__((ext_vector_type(8)))  float    v8f;
typedef __attribute__((ext_vector_type(2)))  float    v2f;

#define NX    32
#define HID   256
#define NOUT  528            // 32 diag + 496 strict-lower
#define BTOT  65536
#define BTILE 128

// ---- branchless transcendental helpers (lower to v_exp_f32/v_log_f32/v_rcp_f32,
// no exec-mask divergence, compiler inserts TRANS32 hazard delays) ----
__device__ inline float fast_tanh(float x) {
  float e2 = __expf(2.f * fabsf(x));                     // v_exp_f32
  float t  = 1.f - 2.f * __builtin_amdgcn_rcpf(e2 + 1.f); // v_rcp_f32
  return copysignf(t, x);                                 // v_bfi_b32
}
__device__ inline float fast_softplus(float x) {
  // stable: softplus(x) = max(x,0) + log(1 + exp(-|x|))
  return fmaxf(x, 0.f) + __logf(1.f + __expf(-fabsf(x)));
}

// ---------------------------------------------------------------------------
// Weight pre-swizzle: f32 row-major [K][N]  ->  f16 WMMA B-fragment order.
// For tile t = kt*NT + nt (32x16 K-by-N tile), lane l, slot j:
//   k = kt*32 + (j>=8 ? 16 : 0) + (l>=16 ? 8 : 0) + (j&7)
//   n = nt*16 + (l & 15)
// ---------------------------------------------------------------------------
__global__ void swz_kernel(const float* __restrict__ W, _Float16* __restrict__ out,
                           int K, int N) {
  int total = K * N;
  int NT = N >> 4;
  for (int idx = blockIdx.x * blockDim.x + threadIdx.x; idx < total;
       idx += gridDim.x * blockDim.x) {
    int t    = idx >> 9;
    int rem  = idx & 511;
    int lane = rem >> 4;
    int j    = rem & 15;
    int kt = t / NT, nt = t % NT;
    int k = kt * 32 + ((j >= 8) ? 16 : 0) + ((lane >= 16) ? 8 : 0) + (j & 7);
    int n = nt * 16 + (lane & 15);
    out[idx] = (_Float16)W[k * N + n];
  }
}

// A fragment (16x32 f16) from row-major LDS tile: lane m<16 holds row m,
// K = 0..7 (slots 0-7) and 16..23 (slots 8-15); lanes 16-31 hold K+8.
__device__ inline v16h ld_afrag_lds(const _Float16* base, int rowstride,
                                    int lane, int kbase) {
  int m  = lane & 15;
  int hv = lane >> 4;
  const _Float16* p = base + m * rowstride + kbase + hv * 8;
  v8h lo = *(const v8h*)(p);        // ds_load_b128
  v8h hi = *(const v8h*)(p + 16);   // ds_load_b128
  v16h r;
#pragma unroll
  for (int i = 0; i < 8; ++i) { r[i] = lo[i]; r[i + 8] = hi[i]; }
  return r;
}

// B fragment from pre-swizzled global weights: one lane = 16 contiguous halves.
__device__ inline v16h ld_bfrag_glb(const _Float16* __restrict__ swz, int t,
                                    int lane) {
  const _Float16* p = swz + (size_t)t * 512 + lane * 16;
  v8h lo = *(const v8h*)(p);        // global_load_b128
  v8h hi = *(const v8h*)(p + 8);    // global_load_b128
  v16h r;
#pragma unroll
  for (int i = 0; i < 8; ++i) { r[i] = lo[i]; r[i + 8] = hi[i]; }
  return r;
}

// ---------------------------------------------------------------------------
// Fused MLP: diag->log -> 3 GEMMs (f16 WMMA, f32 accum) -> nonlinearities ->
// packed L-values (f16) to workspace. 8 waves, each owns a 16-row M block.
// ---------------------------------------------------------------------------
__global__ __launch_bounds__(256, 1) void mlp_kernel(
    const float* __restrict__ P_prev,
    const float* __restrict__ b1, const float* __restrict__ b2,
    const float* __restrict__ b3,
    const _Float16* __restrict__ sw1, const _Float16* __restrict__ sw2,
    const _Float16* __restrict__ sw3,
    _Float16* __restrict__ Lvals) {
  extern __shared__ _Float16 smem[];
  _Float16* a0 = smem;                  // [BTILE][NX]
  _Float16* h1 = a0 + BTILE * NX;       // [BTILE][HID]
  _Float16* h2 = h1 + BTILE * HID;      // [BTILE][HID]

  const int tid  = threadIdx.x;
  const int lane = tid & 31;
  const int wave = tid >> 5;
  const size_t base = (size_t)blockIdx.x * BTILE;

  // ---- stage 0: diagonal extract + log ----
  for (int i = tid; i < BTILE * NX; i += 256) {
    int r = i >> 5, d = i & 31;
    float x = P_prev[(base + r) * (NX * NX) + (size_t)d * (NX + 1)];
    a0[r * NX + d] = (_Float16)__logf(fmaxf(x, 1e-6f));
  }
  __syncthreads();

  // ---- GEMM1: [16x32] @ [32x256] + b1, tanh ----
  {
    v16h afr = ld_afrag_lds(a0 + wave * 16 * NX, NX, lane, 0);
#pragma unroll 4
    for (int nt = 0; nt < HID / 16; ++nt) {
      float bv = b1[nt * 16 + (lane & 15)];
      v8f acc;
#pragma unroll
      for (int i = 0; i < 8; ++i) acc[i] = bv;
      v16h bfr = ld_bfrag_glb(sw1, nt, lane);
      acc = __builtin_amdgcn_wmma_f32_16x16x32_f16(false, afr, false, bfr,
                                                   (short)0, acc, false, false);
      int col = nt * 16 + (lane & 15);
#pragma unroll
      for (int i = 0; i < 8; ++i) {
        int row = wave * 16 + i + ((lane >> 4) << 3);
        h1[row * HID + col] = (_Float16)fast_tanh(acc[i]);
      }
    }
  }
  __syncthreads();

  // ---- GEMM2: [16x256] @ [256x256] + b2, tanh ----
  {
    v16h afr[8];
#pragma unroll
    for (int kt = 0; kt < 8; ++kt)
      afr[kt] = ld_afrag_lds(h1 + wave * 16 * HID, HID, lane, kt * 32);
#pragma unroll 2
    for (int nt = 0; nt < HID / 16; ++nt) {
      float bv = b2[nt * 16 + (lane & 15)];
      v8f acc;
#pragma unroll
      for (int i = 0; i < 8; ++i) acc[i] = bv;
#pragma unroll
      for (int kt = 0; kt < 8; ++kt) {
        v16h bfr = ld_bfrag_glb(sw2, kt * (HID / 16) + nt, lane);
        acc = __builtin_amdgcn_wmma_f32_16x16x32_f16(false, afr[kt], false, bfr,
                                                     (short)0, acc, false, false);
      }
      int col = nt * 16 + (lane & 15);
#pragma unroll
      for (int i = 0; i < 8; ++i) {
        int row = wave * 16 + i + ((lane >> 4) << 3);
        h2[row * HID + col] = (_Float16)fast_tanh(acc[i]);
      }
    }
  }
  __syncthreads();

  // ---- GEMM3: [16x256] @ [256x528] + b3, then diag/lower nonlinearities ----
  {
    v16h afr[8];
#pragma unroll
    for (int kt = 0; kt < 8; ++kt)
      afr[kt] = ld_afrag_lds(h2 + wave * 16 * HID, HID, lane, kt * 32);
    const int NT3 = NOUT / 16;  // 33
#pragma unroll 3
    for (int nt = 0; nt < NT3; ++nt) {
      float bv = b3[nt * 16 + (lane & 15)];
      v8f acc;
#pragma unroll
      for (int i = 0; i < 8; ++i) acc[i] = bv;
#pragma unroll
      for (int kt = 0; kt < 8; ++kt) {
        v16h bfr = ld_bfrag_glb(sw3, kt * NT3 + nt, lane);
        acc = __builtin_amdgcn_wmma_f32_16x16x32_f16(false, afr[kt], false, bfr,
                                                     (short)0, acc, false, false);
      }
      int col = nt * 16 + (lane & 15);
      bool isdiag = (nt < 2);  // columns 0..31 are the diagonal outputs
#pragma unroll
      for (int i = 0; i < 8; ++i) {
        float v = acc[i];
        if (isdiag) {
          v = fminf(fmaxf(fast_softplus(v) + 0.01f, 0.01f), 100.f);
        } else {
          v = 0.1f * fast_tanh(v);
        }
        size_t row = base + wave * 16 + i + ((lane >> 4) << 3);
        Lvals[row * NOUT + col] = (_Float16)v;
      }
    }
  }
}

// ---------------------------------------------------------------------------
// P = L @ L^T + Q per batch element, fp32 WMMA (16x16x4). One batch per wave.
// ---------------------------------------------------------------------------
__global__ __launch_bounds__(256, 1) void llt_kernel(
    const _Float16* __restrict__ Lvals, const float* __restrict__ Q,
    float* __restrict__ P) {
  __shared__ float Lw[8][NX][NX];  // 32 KB, one 32x32 L per wave
  const int lane = threadIdx.x & 31;
  const int wave = threadIdx.x >> 5;
  const size_t b = (size_t)blockIdx.x * 8 + wave;
  const _Float16* lv = Lvals + b * NOUT;

  // lane r builds row r of L: diag at lv[r], strict-lower at lv[32 + r(r-1)/2 + c]
  {
    int r = lane;
    int lowbase = NX + (r * (r - 1)) / 2;
    float* row = &Lw[wave][r][0];
    for (int c = 0; c < NX; ++c) {
      float v = 0.f;
      if (c < r)       v = (float)lv[lowbase + c];
      else if (c == r) v = (float)lv[r];
      row[c] = v;
    }
  }
  __syncthreads();

  // f32 A 16x4 frag: lane l, VGPR v -> (M = l&15, K = 2*(l>>4) + v); B mirrors
  // with N = l&15 and B[k][n] = L[n][k], so both read L rows as float2.
#pragma unroll
  for (int mt = 0; mt < 2; ++mt) {
#pragma unroll
    for (int nt = 0; nt < 2; ++nt) {
      const int n = nt * 16 + (lane & 15);
      v8f acc;
#pragma unroll
      for (int i = 0; i < 8; ++i) {
        int m = mt * 16 + i + ((lane >> 4) << 3);
        acc[i] = Q[m * NX + n];
      }
      const float* pa = &Lw[wave][mt * 16 + (lane & 15)][(lane >> 4) << 1];
      const float* pb = &Lw[wave][nt * 16 + (lane & 15)][(lane >> 4) << 1];
#pragma unroll
      for (int k0 = 0; k0 < 8; ++k0) {
        v2f av = *(const v2f*)(pa + k0 * 4);  // ds_load_b64
        v2f bv = *(const v2f*)(pb + k0 * 4);  // ds_load_b64
        acc = __builtin_amdgcn_wmma_f32_16x16x4_f32(false, av, false, bv,
                                                    (short)0, acc, false, false);
      }
#pragma unroll
      for (int i = 0; i < 8; ++i) {
        int m = mt * 16 + i + ((lane >> 4) << 3);
        P[b * (NX * NX) + m * NX + n] = acc[i];
      }
    }
  }
}

// ---------------------------------------------------------------------------
extern "C" void kernel_launch(void* const* d_in, const int* in_sizes, int n_in,
                              void* d_out, int out_size, void* d_ws, size_t ws_size,
                              hipStream_t stream) {
  const float* P_prev = (const float*)d_in[0];
  const float* W1     = (const float*)d_in[1];
  const float* b1     = (const float*)d_in[2];
  const float* W2     = (const float*)d_in[3];
  const float* b2     = (const float*)d_in[4];
  const float* W3     = (const float*)d_in[5];
  const float* b3     = (const float*)d_in[6];
  const float* Q      = (const float*)d_in[7];
  float* Pout = (float*)d_out;

  // workspace layout (bytes): swizzled f16 weights, then packed L-values
  char* ws = (char*)d_ws;
  _Float16* sw1   = (_Float16*)(ws);                        // 32*256*2   = 16 KB
  _Float16* sw2   = (_Float16*)(ws + 16384);                // 256*256*2  = 128 KB
  _Float16* sw3   = (_Float16*)(ws + 16384 + 131072);       // 256*528*2  = 264 KB
  _Float16* Lvals = (_Float16*)(ws + 16384 + 131072 + 270336);  // B*528*2

  swz_kernel<<<32, 256, 0, stream>>>(W1, sw1, NX, HID);
  swz_kernel<<<256, 256, 0, stream>>>(W2, sw2, HID, HID);
  swz_kernel<<<528, 256, 0, stream>>>(W3, sw3, HID, NOUT);

  const size_t lds_bytes = (size_t)(BTILE * NX + 2 * BTILE * HID) * sizeof(_Float16);
  mlp_kernel<<<BTOT / BTILE, 256, lds_bytes, stream>>>(P_prev, b1, b2, b3,
                                                       sw1, sw2, sw3, Lvals);

  llt_kernel<<<BTOT / 8, 256, 0, stream>>>(Lvals, Q, Pout);
}
